// NeighborhoodAggr_65171833749892
// MI455X (gfx1250) — matrix operand order repaired and checked by reading, not compiled, not run on hardware
//
#include <hip/hip_runtime.h>
#include <math.h>

// Problem constants (from reference)
#define BB    8192
#define MM    64
#define HID   128
#define HEADS 8
#define HD    16
#define TDIM  64

typedef _Float16 v16h __attribute__((ext_vector_type(16)));
typedef _Float16 v8h  __attribute__((ext_vector_type(8)));
typedef float    v8f  __attribute__((ext_vector_type(8)));

// ---- WMMA fragment loaders (CDNA5 16x16x32 f16 layouts, wave32) ----
// A (16x32, f16): lane L holds row M = L%16. Element h of v16h:
//   K = (h/8)*16 + (L/16)*8 + (h%8)   -> two contiguous 8-half chunks.
__device__ __forceinline__ v16h load_a_frag(const _Float16* sA, int mt, int kbase, int lane) {
  int row = mt * 16 + (lane & 15);
  int c   = kbase + ((lane >> 4) << 3);
  v8h lo = *(const v8h*)(sA + row * TDIM + c);
  v8h hi = *(const v8h*)(sA + row * TDIM + c + 16);
  v16h a;
#pragma unroll
  for (int i = 0; i < 8; i++) { a[i] = lo[i]; a[i + 8] = hi[i]; }
  return a;
}

// B (32x16, f16): lane L holds column N = L%16. Element h:
//   K = (h/8)*16 + (L/16)*8 + (h%8)  -> two contiguous 8-half chunks.
// sWt is the weight matrix stored TRANSPOSED in LDS: sWt[n*TDIM + d].
__device__ __forceinline__ v16h load_b_frag(const _Float16* sWt, int nbase, int kbase, int lane) {
  int n = nbase + (lane & 15);
  int k = kbase + ((lane >> 4) << 3);
  v8h lo = *(const v8h*)(sWt + n * TDIM + k);
  v8h hi = *(const v8h*)(sWt + n * TDIM + k + 16);
  v16h x;
#pragma unroll
  for (int i = 0; i < 8; i++) { x[i] = lo[i]; x[i + 8] = hi[i]; }
  return x;
}

// One bias-GEMM phase: sKV[m][n] = (sA @ W)[m][n] + bias[n]  for this wave's
// 16-column tile (nt = wave). A fragments are hoisted so the 8 WMMAs can
// overlap with the LDS loads instead of stalling per-fragment.
__device__ __forceinline__ void wmma_bias_phase(const _Float16* sA, const _Float16* sW,
                                                const float* __restrict__ bias_g,
                                                float* sKV, int wave, int lane) {
  const int nt = wave;                       // 8 waves -> 8 tiles of HID
  v16h b0 = load_b_frag(sW, nt * 16, 0,  lane);
  v16h b1 = load_b_frag(sW, nt * 16, 32, lane);
  v16h a[8];
#pragma unroll
  for (int mt = 0; mt < 4; mt++) {
    a[2 * mt]     = load_a_frag(sA, mt, 0,  lane);
    a[2 * mt + 1] = load_a_frag(sA, mt, 32, lane);
  }
  float bias = bias_g[nt * 16 + (lane & 15)];
  const int ncol  = nt * 16 + (lane & 15);
  const int mbase = (lane >> 4) << 3;
#pragma unroll
  for (int mt = 0; mt < 4; mt++) {
    v8f c;
#pragma unroll
    for (int r = 0; r < 8; r++) c[r] = bias;
    c = __builtin_amdgcn_wmma_f32_16x16x32_f16(false, a[2 * mt],     false, b0, (short)0, c, false, false);
    c = __builtin_amdgcn_wmma_f32_16x16x32_f16(false, a[2 * mt + 1], false, b1, (short)0, c, false, false);
    int mrow = mt * 16 + mbase;
#pragma unroll
    for (int r = 0; r < 8; r++) sKV[(mrow + r) * HID + ncol] = c[r];
  }
}

// Tiny pre-kernel: qbias[n] = (t2v(t) @ wq)[n] + bq[n]  (shared by all B rows)
__global__ void tgat_qbias_kernel(const float* __restrict__ t,
                                  const float* __restrict__ t2v_w,
                                  const float* __restrict__ t2v_b,
                                  const float* __restrict__ wq,
                                  const float* __restrict__ bq,
                                  float* __restrict__ qbias) {
  int n = threadIdx.x;           // 128 threads
  float tt  = t[0];
  float acc = bq[n];
  for (int d = 0; d < TDIM; d++) {
    float raw = tt * t2v_w[d] + t2v_b[d];
    float f   = (d == 0) ? raw : sinf(raw);
    acc += f * wq[d * HID + n];
  }
  qbias[n] = acc;
}

// Main fused kernel: one workgroup per target node b.
__global__ __launch_bounds__(256) void tgat_aggr_kernel(
    const int* __restrict__ nid,   const float* __restrict__ qmat,
    const float* __restrict__ kmat, const float* __restrict__ vmat,
    const int* __restrict__ nbrs,  const float* __restrict__ times,
    const float* __restrict__ tcur,
    const float* __restrict__ t2v_w, const float* __restrict__ t2v_b,
    const float* __restrict__ wk,  const float* __restrict__ bk,
    const float* __restrict__ wv,  const float* __restrict__ bv,
    const float* __restrict__ qbias, float* __restrict__ out) {
  extern __shared__ float4 smem4[];
  char* smem = (char*)smem4;
  _Float16* sA  = (_Float16*)smem;                   // 64x64 f16 time features (A)
  _Float16* sW  = (_Float16*)(smem + 8192);          // 128x64 f16 transposed weights (B)
  float* sKV    = (float*)(smem + 8192 + 16384);     // 64x128 f32 K (then V)
  float* sQ     = sKV + MM * HID;                    // 128
  float* sScore = sQ + HID;                          // 8x64 scores -> exp()
  float* sInv   = sScore + HEADS * MM;               // 8
  float* sTimes = sInv + HEADS;                      // 64
  int*   sMask  = (int*)(sTimes + MM);               // 64

  const int b    = blockIdx.x;
  const int tid  = threadIdx.x;
  const int lane = tid & 31;
  const int wave = tid >> 5;
  const float t0 = tcur[0];

  // ---- stage times + mask ----
  if (tid < MM) {
    float tv = times[b * MM + tid];
    sTimes[tid] = tv;
    sMask[tid]  = (tv <= t0) ? 1 : 0;
  }
  __syncthreads();

  // ---- Time2Vec features (f16) and wk (transposed f16) into LDS ----
  // __sinf -> v_sin_f32: args are O(few units) and results are truncated to
  // f16 anyway, so the native path is the right precision/throughput trade.
  for (int j = 0; j < 16; j++) {              // 4096 elements
    int idx = tid + j * 256;
    int m = idx >> 6, d = idx & 63;
    float raw = sTimes[m] * t2v_w[d] + t2v_b[d];
    float f   = (d == 0) ? raw : __sinf(raw);
    sA[m * TDIM + d] = (_Float16)f;
  }
  for (int j = 0; j < 32; j++) {              // 8192 elements, coalesced read
    int idx = tid + j * 256;
    int d = idx >> 7, n = idx & 127;
    sW[n * TDIM + d] = (_Float16)wk[idx];
  }
  __syncthreads();

  // ---- WMMA: K time-bias = sA @ wk + bk  -> sKV ----
  wmma_bias_phase(sA, sW, bk, sKV, wave, lane);
  __syncthreads();

  // ---- gather kmat rows (float4 coalesced) and add; also q ----
#pragma unroll
  for (int i = 0; i < 8; i++) {
    int m   = wave * 8 + i;
    int nbr = nbrs[b * MM + m];
    float4 g = ((const float4*)(kmat + (long)nbr * HID))[lane];
    float* p = sKV + m * HID + lane * 4;
    p[0] += g.x; p[1] += g.y; p[2] += g.z; p[3] += g.w;
  }
  if (tid < HID) sQ[tid] = qmat[(long)nid[b] * HID + tid] + qbias[tid];
  __syncthreads();

  // ---- scores[h][m] = 0.25 * q_h . k_{m,h}, masked ----
  for (int j = 0; j < 2; j++) {
    int p = tid + j * 256;                    // 512 (h,m) pairs
    int h = p >> 6, m = p & 63;
    float acc = 0.f;
#pragma unroll
    for (int d = 0; d < HD; d++)
      acc += sQ[h * HD + d] * sKV[m * HID + h * HD + d];
    sScore[h * MM + m] = sMask[m] ? acc * 0.25f : -3.402823466e38f;
  }
  __syncthreads();

  // ---- masked softmax per head (tiny; 8 lanes) ----
  if (tid < HEADS) {
    int h = tid;
    float mx = -3.402823466e38f;
    int any = 0;
    for (int m = 0; m < MM; m++)
      if (sMask[m]) { any = 1; mx = fmaxf(mx, sScore[h * MM + m]); }
    float sum = 0.f;
    for (int m = 0; m < MM; m++) {
      float e = sMask[m] ? __expf(sScore[h * MM + m] - mx) : 0.f;
      sScore[h * MM + m] = e;
      sum += e;
    }
    sInv[h] = (any && sum > 0.f) ? (1.0f / sum) : 0.f;
  }
  __syncthreads();

  // ---- reuse sW for wv (K path fully consumed) ----
  for (int j = 0; j < 32; j++) {
    int idx = tid + j * 256;
    int d = idx >> 7, n = idx & 127;
    sW[n * TDIM + d] = (_Float16)wv[idx];
  }
  __syncthreads();

  // ---- WMMA: V time-bias = sA @ wv + bv -> sKV (overwrite K) ----
  wmma_bias_phase(sA, sW, bv, sKV, wave, lane);
  __syncthreads();

  // ---- gather vmat rows and add ----
#pragma unroll
  for (int i = 0; i < 8; i++) {
    int m   = wave * 8 + i;
    int nbr = nbrs[b * MM + m];
    float4 g = ((const float4*)(vmat + (long)nbr * HID))[lane];
    float* p = sKV + m * HID + lane * 4;
    p[0] += g.x; p[1] += g.y; p[2] += g.z; p[3] += g.w;
  }
  __syncthreads();

  // ---- out[b][n] = sum_m attn[h][m] * v[m][n]   (h = n/16) ----
  if (tid < HID) {
    int h = tid >> 4;
    float acc = 0.f;
    for (int m = 0; m < MM; m++)
      acc += sScore[h * MM + m] * sKV[m * HID + tid];
    out[(long)b * HID + tid] = acc * sInv[h];
  }
}

extern "C" void kernel_launch(void* const* d_in, const int* in_sizes, int n_in,
                              void* d_out, int out_size, void* d_ws, size_t ws_size,
                              hipStream_t stream) {
  const int*   nid   = (const int*)  d_in[0];
  const float* qmat  = (const float*)d_in[1];
  const float* kmat  = (const float*)d_in[2];
  const float* vmat  = (const float*)d_in[3];
  const int*   nbrs  = (const int*)  d_in[4];
  const float* times = (const float*)d_in[5];
  const float* t     = (const float*)d_in[6];
  const float* t2vw  = (const float*)d_in[7];
  const float* t2vb  = (const float*)d_in[8];
  const float* wk    = (const float*)d_in[9];
  const float* bk    = (const float*)d_in[10];
  const float* wq    = (const float*)d_in[11];
  const float* bq    = (const float*)d_in[12];
  const float* wv    = (const float*)d_in[13];
  const float* bv    = (const float*)d_in[14];
  float* out   = (float*)d_out;
  float* qbias = (float*)d_ws;   // 128 floats

  tgat_qbias_kernel<<<1, 128, 0, stream>>>(t, t2vw, t2vb, wq, bq, qbias);

  // LDS: sA(8192) + sW(16384) + sKV(32768) + sQ(512) + sScore(2048)
  //      + sInv(32) + sTimes(256) + sMask(256) = 60448 bytes
  size_t smem = 8192 + 16384 + (size_t)MM * HID * 4 + HID * 4 +
                HEADS * MM * 4 + HEADS * 4 + MM * 4 + MM * 4;
  tgat_aggr_kernel<<<BB, 256, smem, stream>>>(nid, qmat, kmat, vmat, nbrs, times,
                                              t, t2vw, t2vb, wk, bk, wv, bv,
                                              qbias, out);
}
